// GPTNeoGQASelfAttention_50122268344701
// MI455X (gfx1250) — matrix operand order repaired and checked by reading, not compiled, not run on hardware
//
#include <hip/hip_runtime.h>
#include <hip/hip_bf16.h>

typedef __bf16 bf16;
typedef __attribute__((ext_vector_type(16))) __bf16 v16bf;
typedef __attribute__((ext_vector_type(8)))  __bf16 v8bf;
typedef __attribute__((ext_vector_type(8)))  float  v8f;
typedef __attribute__((ext_vector_type(4)))  int    v4i;

#define S_LEN  2048
#define E_DIM  2048
#define D_HEAD 128
#define KVDIM  512      // 4 kv heads * 128
#define WIN    256
#define NEG_F  (-3.0e38f)

#if __has_builtin(__builtin_amdgcn_global_load_async_to_lds_b128) && \
    __has_builtin(__builtin_amdgcn_s_wait_asynccnt)
#define HAVE_ASYNC_LDS 1
#else
#define HAVE_ASYNC_LDS 0
#endif

// ---------- helpers ----------
static __device__ __forceinline__ bf16 f2b(float f) {
  unsigned u = __builtin_bit_cast(unsigned, f);
  unsigned r = u + 0x7FFFu + ((u >> 16) & 1u);      // round-to-nearest-even
  unsigned short h = (unsigned short)(r >> 16);
  return __builtin_bit_cast(bf16, h);
}

static __device__ __forceinline__ v8f wmma_bf16(v16bf a, v16bf b, v8f c) {
  // (neg_a, A, neg_b, B, c_mod, C, reuse_a, reuse_b)
  return __builtin_amdgcn_wmma_f32_16x16x32_bf16(false, a, false, b, (short)0, c, false, false);
}

// A fragment (16x32 bf16): lane holds row = lane&15; elems 0..15 = K(k0+off .. +15),
// off = (lane<16) ? 0 : 16.
static __device__ __forceinline__ v16bf ld_afrag(const bf16* base, int stride, long row, int k0, int lane) {
  const bf16* p = base + row * (long)stride + k0 + ((lane & 16) ? 16 : 0);
  v8bf lo = *(const v8bf*)(p);
  v8bf hi = *(const v8bf*)(p + 8);
  return __builtin_shufflevector(lo, hi, 0,1,2,3,4,5,6,7,8,9,10,11,12,13,14,15);
}

// B fragment (32x16 bf16, B[k][n] = W[n][k]): lane holds col = lane&15;
// elems 0..15 = K(k0+koff .. +15), koff = (lane<16) ? 0 : 16  -> one 32B load.
static __device__ __forceinline__ v16bf ld_bfrag(const bf16* base, int stride, long row, int k0, int lane) {
  const bf16* p = base + row * (long)stride + k0 + ((lane & 16) ? 16 : 0);
  return *(const v16bf*)(p);
}

// ---------- conversion kernels ----------
__global__ void k_f32_to_bf16(const float* __restrict__ src, bf16* __restrict__ dst, long n) {
  long i = (long)blockIdx.x * blockDim.x + threadIdx.x;
  long stride = (long)gridDim.x * blockDim.x;
  for (; i < n; i += stride) dst[i] = f2b(src[i]);
}

// Wo_red[e, g*128+d] = sum_{r<4} Wo[e, (4g+r)*128+d]   (exploits duplicated head groups)
__global__ void k_fold_wo(const float* __restrict__ Wo, bf16* __restrict__ Wr, long n) {
  long i = (long)blockIdx.x * blockDim.x + threadIdx.x;
  long stride = (long)gridDim.x * blockDim.x;
  for (; i < n; i += stride) {
    long e = i / KVDIM;
    int  c = (int)(i % KVDIM);
    int  g = c >> 7, d = c & 127;
    const float* p = Wo + e * (long)E_DIM + (long)(g * 4) * D_HEAD + d;
    Wr[i] = f2b(p[0] + p[128] + p[256] + p[384]);
  }
}

// ---------- GEMM: O[M,N] = A[M,K] @ W[N,K]^T ----------
// Each wave owns a 16x64 strip: A-frag loaded once per k-step, reused for 4 WMMAs.
__global__ __launch_bounds__(256) void k_gemm_bf16out(const bf16* __restrict__ A, const bf16* __restrict__ W,
                                                      bf16* __restrict__ O, int N, int K) {
  int lane = threadIdx.x & 31, wave = threadIdx.x >> 5;
  int  n0 = blockIdx.x * 64;
  long m0 = (long)blockIdx.y * 128 + wave * 16;
  long arow = m0 + (lane & 15);
  long wrow = (long)n0 + (lane & 15);
  v8f acc[4] = {};
  for (int k = 0; k < K; k += 32) {
    if (k + 64 < K) {                 // pull next cachelines toward the WGP
      __builtin_prefetch(A + arow * K + k + 64, 0, 1);
      __builtin_prefetch(W + wrow * K + k + 64, 0, 1);
    }
    v16bf a = ld_afrag(A, K, arow, k, lane);
#pragma unroll
    for (int t = 0; t < 4; ++t) {
      v16bf b = ld_bfrag(W, K, wrow + t * 16, k, lane);
      acc[t] = wmma_bf16(a, b, acc[t]);
    }
  }
  int col = lane & 15, half = lane >> 4;
#pragma unroll
  for (int t = 0; t < 4; ++t)
#pragma unroll
    for (int r = 0; r < 8; ++r) {
      long row = m0 + r + half * 8;        // C/D layout: VGPR r -> rows r / r+8
      O[row * N + n0 + t * 16 + col] = f2b(acc[t][r]);
    }
}

__global__ __launch_bounds__(256) void k_gemm_f32out_bias(const bf16* __restrict__ A, const bf16* __restrict__ W,
                                                          const float* __restrict__ bias, float* __restrict__ O,
                                                          int N, int K) {
  int lane = threadIdx.x & 31, wave = threadIdx.x >> 5;
  int  n0 = blockIdx.x * 64;
  long m0 = (long)blockIdx.y * 128 + wave * 16;
  long arow = m0 + (lane & 15);
  long wrow = (long)n0 + (lane & 15);
  v8f acc[4] = {};
  for (int k = 0; k < K; k += 32) {
    if (k + 64 < K) {
      __builtin_prefetch(A + arow * K + k + 64, 0, 1);
      __builtin_prefetch(W + wrow * K + k + 64, 0, 1);
    }
    v16bf a = ld_afrag(A, K, arow, k, lane);
#pragma unroll
    for (int t = 0; t < 4; ++t) {
      v16bf b = ld_bfrag(W, K, wrow + t * 16, k, lane);
      acc[t] = wmma_bf16(a, b, acc[t]);
    }
  }
  int col = lane & 15, half = lane >> 4;
#pragma unroll
  for (int t = 0; t < 4; ++t) {
    float bv = bias[n0 + t * 16 + col];
#pragma unroll
    for (int r = 0; r < 8; ++r) {
      long row = m0 + r + half * 8;
      O[row * N + n0 + t * 16 + col] = acc[t][r] + bv;
    }
  }
}

// ---------- flash-style sliding-window attention ----------
// grid = B * HKV * (S/16) single-wave blocks; one wave owns a 16-query tile of one kv-head.
__global__ __launch_bounds__(32) void k_attn(const bf16* __restrict__ Qb, const bf16* __restrict__ Kb,
                                             const bf16* __restrict__ Vb, bf16* __restrict__ Ab) {
  __shared__ alignas(32) bf16 Vs[32][128];   // staged V block (32 keys x 128 d)
  __shared__ alignas(32) bf16 Ps[16][32];    // P tile for D-layout -> A-layout conversion
  int lane = threadIdx.x;
  int col = lane & 15, half = lane >> 4;
  int tile = blockIdx.x;
  int qt = tile & 127;
  int g  = (tile >> 7) & 3;
  int b  = tile >> 9;
  long base = (long)b * S_LEN;
  int q0 = qt * 16;

  v16bf qf[4];
#pragma unroll
  for (int c = 0; c < 4; ++c)
    qf[c] = ld_afrag(Qb, KVDIM, base + q0 + (lane & 15), g * D_HEAD + c * 32, lane);

  float mrow[8], lrow[8];
  v8f o[8] = {};
#pragma unroll
  for (int r = 0; r < 8; ++r) { mrow[r] = NEG_F; lrow[r] = 0.f; }

  int jmin = q0 - (WIN - 1); if (jmin < 0) jmin = 0;
  int jlo = jmin & ~31;
  int jhi = (q0 + 15) & ~31;

  for (int j0 = jlo; j0 <= jhi; j0 += 32) {
    { // stage V rows, one row per lane (async DMA to LDS when available)
      const bf16* src = Vb + (base + j0 + lane) * KVDIM + g * D_HEAD;
      bf16* dst = &Vs[lane][0];
#if HAVE_ASYNC_LDS
#pragma unroll
      for (int c = 0; c < 16; ++c)
        __builtin_amdgcn_global_load_async_to_lds_b128(
            (v4i*)(src + c * 8), (v4i*)(dst + c * 8), 0, 0);
      __builtin_amdgcn_s_wait_asynccnt(0);
#else
#pragma unroll
      for (int c = 0; c < 8; ++c)
        *(v16bf*)&dst[c * 16] = *(const v16bf*)(src + c * 16);
#endif
    }
    __syncthreads();

    // scores: two 16x16 tiles, K-dim 128 = 4 x WMMA each
    float sc[2][8];
#pragma unroll
    for (int nc = 0; nc < 2; ++nc) {
      v8f acc = {};
#pragma unroll
      for (int c = 0; c < 4; ++c) {
        v16bf kf = ld_bfrag(Kb, KVDIM, base + j0 + nc * 16 + col, g * D_HEAD + c * 32, lane);
        acc = wmma_bf16(qf[c], kf, acc);
      }
      int j = j0 + nc * 16 + col;
#pragma unroll
      for (int r = 0; r < 8; ++r) {
        int i = q0 + r + half * 8;
        bool ok = (j <= i) && (j > i - WIN);
        sc[nc][r] = ok ? acc[r] : NEG_F;
      }
    }

    // online softmax update (row stats live in the 16 lanes of each half)
#pragma unroll
    for (int r = 0; r < 8; ++r) {
      float bm = fmaxf(sc[0][r], sc[1][r]);
#pragma unroll
      for (int msk = 8; msk >= 1; msk >>= 1) bm = fmaxf(bm, __shfl_xor(bm, msk, 32));
      float mn = fmaxf(mrow[r], bm);
      float sca = (mrow[r] > -1.0e37f) ? __expf(mrow[r] - mn) : 0.f;
      float p0 = (sc[0][r] > -1.0e37f) ? __expf(sc[0][r] - mn) : 0.f;
      float p1 = (sc[1][r] > -1.0e37f) ? __expf(sc[1][r] - mn) : 0.f;
      float rs = p0 + p1;
#pragma unroll
      for (int msk = 8; msk >= 1; msk >>= 1) rs += __shfl_xor(rs, msk, 32);
      lrow[r] = lrow[r] * sca + rs;
      mrow[r] = mn;
      Ps[r + half * 8][col]      = f2b(p0);
      Ps[r + half * 8][16 + col] = f2b(p1);
#pragma unroll
      for (int d = 0; d < 8; ++d) o[d][r] *= sca;
    }
    __syncthreads();

    // P @ V : A = P(16x32) from LDS, B = V-block(32x16) per 16-wide d-chunk
    v16bf pf = ld_afrag(&Ps[0][0], 32, lane & 15, 0, lane);
    int koff = half * 16;
#pragma unroll
    for (int d = 0; d < 8; ++d) {
      v16bf vf;
#pragma unroll
      for (int e = 0; e < 16; ++e) vf[e] = Vs[koff + e][d * 16 + col];
      o[d] = wmma_bf16(pf, vf, o[d]);
    }
    __syncthreads();
  }

  // normalize and emit compact (B*S, 512) context (one copy per kv-head group)
#pragma unroll
  for (int r = 0; r < 8; ++r) {
    long row = base + q0 + r + half * 8;
    float inv = 1.0f / lrow[r];
#pragma unroll
    for (int d = 0; d < 8; ++d)
      Ab[row * KVDIM + g * D_HEAD + d * 16 + col] = f2b(o[d][r] * inv);
  }
}

// ---------- launch ----------
extern "C" void kernel_launch(void* const* d_in, const int* in_sizes, int n_in,
                              void* d_out, int out_size, void* d_ws, size_t ws_size,
                              hipStream_t stream) {
  const float* x  = (const float*)d_in[0];
  const float* Wq = (const float*)d_in[1];
  const float* Wk = (const float*)d_in[2];
  const float* Wv = (const float*)d_in[3];
  const float* Wo = (const float*)d_in[4];
  const float* bo = (const float*)d_in[5];

  const long M = 2L * S_LEN;           // B*S = 4096
  char* ws = (char*)d_ws;
  size_t off = 0;
  auto alloc = [&](size_t bytes) -> char* {
    char* p = ws + off;
    off += (bytes + 255) & ~(size_t)255;
    return p;
  };
  bf16* xb  = (bf16*)alloc((size_t)M * E_DIM * 2);
  bf16* wqb = (bf16*)alloc((size_t)KVDIM * E_DIM * 2);   // only q-heads 0..3 are ever used
  bf16* wkb = (bf16*)alloc((size_t)KVDIM * E_DIM * 2);
  bf16* wvb = (bf16*)alloc((size_t)KVDIM * E_DIM * 2);
  bf16* wrb = (bf16*)alloc((size_t)E_DIM * KVDIM * 2);   // folded Wo
  bf16* Qb  = (bf16*)alloc((size_t)M * KVDIM * 2);
  bf16* Kb  = (bf16*)alloc((size_t)M * KVDIM * 2);
  bf16* Vb  = (bf16*)alloc((size_t)M * KVDIM * 2);
  bf16* Ab  = (bf16*)alloc((size_t)M * KVDIM * 2);

  k_f32_to_bf16<<<2048, 256, 0, stream>>>(x,  xb,  M * (long)E_DIM);
  k_f32_to_bf16<<<1024, 256, 0, stream>>>(Wq, wqb, (long)KVDIM * E_DIM); // first 512 rows of Wq
  k_f32_to_bf16<<<1024, 256, 0, stream>>>(Wk, wkb, (long)KVDIM * E_DIM);
  k_f32_to_bf16<<<1024, 256, 0, stream>>>(Wv, wvb, (long)KVDIM * E_DIM);
  k_fold_wo    <<<1024, 256, 0, stream>>>(Wo, wrb, (long)E_DIM * KVDIM);

  dim3 blk(256);
  k_gemm_bf16out<<<dim3(KVDIM / 64, M / 128), blk, 0, stream>>>(xb, wqb, Qb, KVDIM, E_DIM);
  k_gemm_bf16out<<<dim3(KVDIM / 64, M / 128), blk, 0, stream>>>(xb, wkb, Kb, KVDIM, E_DIM);
  k_gemm_bf16out<<<dim3(KVDIM / 64, M / 128), blk, 0, stream>>>(xb, wvb, Vb, KVDIM, E_DIM);

  k_attn<<<dim3(2 * 4 * (S_LEN / 16)), dim3(32), 0, stream>>>(Qb, Kb, Vb, Ab);

  k_gemm_f32out_bias<<<dim3(E_DIM / 64, M / 128), blk, 0, stream>>>(Ab, wrb, bo, (float*)d_out, E_DIM, KVDIM);
}
